// Decoder_42485816492413
// MI455X (gfx1250) — matrix operand order repaired and checked by reading, not compiled, not run on hardware
//
#include <hip/hip_runtime.h>

// ---------------------------------------------------------------------------
// Tacotron2-style decoder for MI455X (gfx1250, wave32, WMMA).
// bf16 weights (~37MB, L2-resident), V_WMMA_F32_16X16X32_BF16 for all GEMMs
// (batch M=16 == one WMMA tile), branch-free region-split K loops, fused
// lane-local LSTM pointwise updates, parallel attention kernels.
// ---------------------------------------------------------------------------

typedef __attribute__((ext_vector_type(8)))  __bf16 v8bf;
typedef __attribute__((ext_vector_type(16))) __bf16 v16bf;
typedef __attribute__((ext_vector_type(8)))  float  v8f;

#define B_      16
#define T_ENC   200
#define T_DEC   250
#define N_MELS  80
#define PRENET_ 256
#define ENC_H   512
#define ATTN_H  1024
#define DEC_H   1024
#define ATTN_D  128
#define N_FILT  32
#define KSIZE   31
#define PAD_    15
#define KA      1792      // prenet(256)+ctx(512)+ah(1024)
#define KD      2560      // ctx(512)+ah(1024)+dh(1024)
#define KP      1536      // dh(1024)+ctx(512)
#define GA      4096      // 4*hidden (gate rows)

__device__ inline unsigned short f2bf(float f) {          // RNE f32->bf16
  unsigned u = __float_as_uint(f);
  unsigned r = (u + 0x7FFFu + ((u >> 16) & 1u)) >> 16;
  return (unsigned short)r;
}
__device__ inline float sigm(float x) { return 1.0f / (1.0f + __expf(-x)); }

__device__ inline v16bf cat8(v8bf lo, v8bf hi) {
  return __builtin_shufflevector(lo, hi, 0,1,2,3,4,5,6,7,8,9,10,11,12,13,14,15);
}
#define WMMA_BF16(A, Bv, C) \
  __builtin_amdgcn_wmma_f32_16x16x32_bf16(false, (A), false, (Bv), (short)0, (C), false, false)

// ---------------------------------------------------------------------------
// Region GEMM helpers. Pointers are pre-offset per lane:
//   xa = Xregion + m*stride + (hi?16:0)     (A 16x32 frag = 16 contiguous halfs)
//   w* = Wrow(n) + colBase  + (hi?16:0)     (B 32x16 frag = 16 contiguous halfs)
// One-chunk software pipeline: loads for chunk k+1 issue before WMMAs of k.
// ---------------------------------------------------------------------------
__device__ inline void gemm4_region(const unsigned short* xa, int L,
    const unsigned short* w0, const unsigned short* w1,
    const unsigned short* w2, const unsigned short* w3,
    v8f& gi, v8f& gf, v8f& gg, v8f& go) {
  v16bf a  = *(const v16bf*)(const void*)xa;
  v16bf b0 = *(const v16bf*)(const void*)w0;
  v16bf b1 = *(const v16bf*)(const void*)w1;
  v16bf b2 = *(const v16bf*)(const void*)w2;
  v16bf b3 = *(const v16bf*)(const void*)w3;
  for (int kc = 32; kc < L; kc += 32) {
    v16bf na  = *(const v16bf*)(const void*)(xa + kc);
    v16bf nb0 = *(const v16bf*)(const void*)(w0 + kc);
    v16bf nb1 = *(const v16bf*)(const void*)(w1 + kc);
    v16bf nb2 = *(const v16bf*)(const void*)(w2 + kc);
    v16bf nb3 = *(const v16bf*)(const void*)(w3 + kc);
    gi = WMMA_BF16(a, b0, gi);
    gf = WMMA_BF16(a, b1, gf);
    gg = WMMA_BF16(a, b2, gg);
    go = WMMA_BF16(a, b3, go);
    a = na; b0 = nb0; b1 = nb1; b2 = nb2; b3 = nb3;
  }
  gi = WMMA_BF16(a, b0, gi);
  gf = WMMA_BF16(a, b1, gf);
  gg = WMMA_BF16(a, b2, gg);
  go = WMMA_BF16(a, b3, go);
}

__device__ inline void gemm1_region(const unsigned short* xa, int L,
                                    const unsigned short* w0, v8f& acc) {
  v16bf a  = *(const v16bf*)(const void*)xa;
  v16bf b0 = *(const v16bf*)(const void*)w0;
  for (int kc = 32; kc < L; kc += 32) {
    v16bf na  = *(const v16bf*)(const void*)(xa + kc);
    v16bf nb0 = *(const v16bf*)(const void*)(w0 + kc);
    acc = WMMA_BF16(a, b0, acc);
    a = na; b0 = nb0;
  }
  acc = WMMA_BF16(a, b0, acc);
}

// ---------------- one-time prep kernels ----------------

__global__ __launch_bounds__(256) void k_prep_weights(
    const float* awih, const float* awhh, const float* abih, const float* abhh,
    const float* dwih, const float* dwhh, const float* dbih, const float* dbhh,
    const float* pjw, const float* qw,
    unsigned short* Wa, unsigned short* Wd, unsigned short* Wp,
    unsigned short* Qwb, float* ba, float* bd) {
  const long long NA = (long long)GA * KA, ND = (long long)GA * KD;
  const long long NP = (long long)N_MELS * KP, NQ = (long long)ATTN_D * ATTN_H;
  const long long total = NA + ND + NP + NQ + GA + GA;
  for (long long i = (long long)blockIdx.x * blockDim.x + threadIdx.x; i < total;
       i += (long long)gridDim.x * blockDim.x) {
    if (i < NA) {
      int n = (int)(i / KA), k = (int)(i % KA);
      float v = (k < 768) ? awih[(size_t)n * 768 + k] : awhh[(size_t)n * 1024 + (k - 768)];
      Wa[i] = f2bf(v);
    } else if (i < NA + ND) {
      long long j = i - NA; int n = (int)(j / KD), k = (int)(j % KD);
      float v = (k < 1536) ? dwih[(size_t)n * 1536 + k] : dwhh[(size_t)n * 1024 + (k - 1536)];
      Wd[j] = f2bf(v);
    } else if (i < NA + ND + NP) {
      long long j = i - NA - ND;
      Wp[j] = f2bf(pjw[j]);
    } else if (i < NA + ND + NP + NQ) {
      long long j = i - NA - ND - NP;
      Qwb[j] = f2bf(qw[j]);
    } else if (i < NA + ND + NP + NQ + GA) {
      int n = (int)(i - (NA + ND + NP + NQ));
      ba[n] = abih[n] + abhh[n];
    } else {
      int n = (int)(i - (NA + ND + NP + NQ + GA));
      bd[n] = dbih[n] + dbhh[n];
    }
  }
}

// prenet over shifted targets -> bf16, layout [t][b][j]
__global__ __launch_bounds__(256) void k_prenet(const float* tgt, const float* pw,
                                                const float* pb, unsigned short* pre) {
  int idx = blockIdx.x * blockDim.x + threadIdx.x;
  if (idx >= T_DEC * B_ * PRENET_) return;
  int j = idx & 255, b = (idx >> 8) & 15, t = idx >> 12;
  float s = pb[j];
  if (t > 0) {
    const float* x = tgt + ((size_t)b * T_DEC + (t - 1)) * N_MELS;
    const float* w = pw + (size_t)j * N_MELS;
    for (int k = 0; k < N_MELS; ++k) s += x[k] * w[k];
  }
  pre[idx] = f2bf(fmaxf(s, 0.0f));
}

// pme[b][t][d] = enc[b][t] . M_w[d]
__global__ __launch_bounds__(256) void k_pme(const float* enc, const float* Mw, float* pme) {
  int idx = blockIdx.x * blockDim.x + threadIdx.x;
  if (idx >= B_ * T_ENC * ATTN_D) return;
  int d = idx & 127, bt = idx >> 7;
  const float* er = enc + (size_t)bt * ENC_H;
  const float* mr = Mw + (size_t)d * ENC_H;
  float s = 0.0f;
  for (int k = 0; k < ENC_H; ++k) s += er[k] * mr[k];
  pme[idx] = s;
}

// ---------------- per-step kernels ----------------

// Fused attn-LSTM: 3-region WMMA gates GEMM + lane-local cell update.
__global__ __launch_bounds__(128) void k_attn_lstm(
    const unsigned short* Wa, const float* ba,
    const unsigned short* pre_t, const unsigned short* ctxb,
    const unsigned short* ahb_prev, unsigned short* ahb_new, float* ac) {
  int  wid  = blockIdx.x * (blockDim.x >> 5) + (threadIdx.x >> 5);   // 0..63
  int  j0   = wid * 16;
  int  lane = threadIdx.x & 31;
  bool hi   = lane >= 16;
  int  mn   = lane & 15;
  int  aoff = hi ? 16 : 0, boff = hi ? 16 : 0;
  const unsigned short* w0 = Wa + (size_t)(j0 + mn) * KA + boff;
  const unsigned short* w1 = w0 + (size_t)1024 * KA;
  const unsigned short* w2 = w0 + (size_t)2048 * KA;
  const unsigned short* w3 = w0 + (size_t)3072 * KA;
  v8f gi = {}, gf = {}, gg = {}, go = {};
  gemm4_region(pre_t    + (size_t)mn * PRENET_ + aoff, 256,
               w0,       w1,       w2,       w3,       gi, gf, gg, go);
  gemm4_region(ctxb     + (size_t)mn * ENC_H   + aoff, 512,
               w0 + 256, w1 + 256, w2 + 256, w3 + 256, gi, gf, gg, go);
  gemm4_region(ahb_prev + (size_t)mn * ATTN_H  + aoff, 1024,
               w0 + 768, w1 + 768, w2 + 768, w3 + 768, gi, gf, gg, go);
  int j = j0 + mn, mb = hi ? 8 : 0;
  float bi = ba[j], bff = ba[j + 1024], bg = ba[j + 2048], bo = ba[j + 3072];
  for (int r = 0; r < 8; ++r) {
    int m = mb + r;
    float iv = sigm(gi[r] + bi);
    float fv = sigm(gf[r] + bff);
    float gv = tanhf(gg[r] + bg);
    float ov = sigm(go[r] + bo);
    float c2 = fv * ac[m * ATTN_H + j] + iv * gv;
    float h2 = ov * tanhf(c2);
    ac[m * ATTN_H + j] = c2;
    ahb_new[m * ATTN_H + j] = f2bf(h2);
  }
}

// q = ah @ Q_w^T  (16x1024 @ 1024x128) via WMMA: 8 waves, one N-tile each.
__global__ __launch_bounds__(256) void k_q(const unsigned short* Qwb,
                                           const unsigned short* ahb, float* q) {
  int  n0   = (threadIdx.x >> 5) * 16;
  int  lane = threadIdx.x & 31;
  bool hi   = lane >= 16;
  int  mn   = lane & 15;
  const unsigned short* xa = ahb + (size_t)mn * ATTN_H + (hi ? 16 : 0);
  const unsigned short* wr = Qwb + (size_t)(n0 + mn) * ATTN_H + (hi ? 16 : 0);
  v8f acc = {};
  gemm1_region(xa, ATTN_H, wr, acc);
  for (int r = 0; r < 8; ++r) {
    int m = (hi ? 8 : 0) + r;
    q[m * ATTN_D + n0 + mn] = acc[r];
  }
}

// One block per (b,t): conv taps -> lc[32] in LDS, 128 threads do d-dots+tanh,
// tree-reduce to a single energy.
__global__ __launch_bounds__(128) void k_energy(
    const float* q, const float* pme, const float* aw, const float* aws,
    const float* Lw, const float* Ww, const float* convw, float* e) {
  int b = blockIdx.x / T_ENC, tt = blockIdx.x % T_ENC;
  __shared__ float lc[N_FILT];
  __shared__ float part[128];
  int tid = threadIdx.x;
  if (tid < N_FILT) {
    float s = 0.0f;
    for (int k = 0; k < KSIZE; ++k) {
      int p = tt + k - PAD_;
      if (0 <= p && p < T_ENC)
        s += aw[b * T_ENC + p] * convw[tid * 62 + k]
           + aws[b * T_ENC + p] * convw[tid * 62 + 31 + k];
    }
    lc[tid] = s;
  }
  __syncthreads();
  int d = tid;                                   // blockDim == ATTN_D
  float ld = 0.0f;
  for (int f = 0; f < N_FILT; ++f) ld += Lw[d * N_FILT + f] * lc[f];
  part[tid] = Ww[d] * tanhf(q[b * ATTN_D + d] + pme[((size_t)b * T_ENC + tt) * ATTN_D + d] + ld);
  __syncthreads();
  for (int s = 64; s > 0; s >>= 1) {
    if (tid < s) part[tid] += part[tid + s];
    __syncthreads();
  }
  if (tid == 0) e[b * T_ENC + tt] = part[0];
}

// One block per batch row: parallel softmax + context, bf16 ctx out.
__global__ __launch_bounds__(256) void k_softmax_ctx(
    const float* e, const float* enc, float* aw, float* aws,
    unsigned short* ctxb, float* align_out, int t) {
  int b = blockIdx.x;
  __shared__ float sw[T_ENC];
  __shared__ float red[256];
  int tid = threadIdx.x;
  float v = (tid < T_ENC) ? e[b * T_ENC + tid] : -1e30f;
  red[tid] = v;
  __syncthreads();
  for (int s = 128; s > 0; s >>= 1) {
    if (tid < s) red[tid] = fmaxf(red[tid], red[tid + s]);
    __syncthreads();
  }
  float mx = red[0];
  __syncthreads();
  float ex = (tid < T_ENC) ? __expf(v - mx) : 0.0f;
  red[tid] = ex;
  __syncthreads();
  for (int s = 128; s > 0; s >>= 1) {
    if (tid < s) red[tid] += red[tid + s];
    __syncthreads();
  }
  float inv = 1.0f / red[0];
  if (tid < T_ENC) {
    float w = ex * inv;
    sw[tid] = w;
    aw[b * T_ENC + tid] = w;
    aws[b * T_ENC + tid] += w;
    align_out[((size_t)b * T_DEC + t) * T_ENC + tid] = w;
  }
  __syncthreads();
  for (int ec = tid; ec < ENC_H; ec += 256) {
    const float* er = enc + ((size_t)b * T_ENC) * ENC_H + ec;
    float s = 0.0f;
    for (int tt = 0; tt < T_ENC; ++tt) s += sw[tt] * er[(size_t)tt * ENC_H];
    ctxb[b * ENC_H + ec] = f2bf(s);
  }
}

// Fused decoder-LSTM: 3-region WMMA gates GEMM + cell update.
__global__ __launch_bounds__(128) void k_dec_lstm(
    const unsigned short* Wd, const float* bd,
    const unsigned short* ctxb, const unsigned short* ahb_new,
    const unsigned short* dhb_prev, unsigned short* dhb_new, float* dc) {
  int  wid  = blockIdx.x * (blockDim.x >> 5) + (threadIdx.x >> 5);
  int  j0   = wid * 16;
  int  lane = threadIdx.x & 31;
  bool hi   = lane >= 16;
  int  mn   = lane & 15;
  int  aoff = hi ? 16 : 0, boff = hi ? 16 : 0;
  const unsigned short* w0 = Wd + (size_t)(j0 + mn) * KD + boff;
  const unsigned short* w1 = w0 + (size_t)1024 * KD;
  const unsigned short* w2 = w0 + (size_t)2048 * KD;
  const unsigned short* w3 = w0 + (size_t)3072 * KD;
  v8f gi = {}, gf = {}, gg = {}, go = {};
  gemm4_region(ctxb     + (size_t)mn * ENC_H  + aoff, 512,
               w0,        w1,        w2,        w3,        gi, gf, gg, go);
  gemm4_region(ahb_new  + (size_t)mn * ATTN_H + aoff, 1024,
               w0 + 512,  w1 + 512,  w2 + 512,  w3 + 512,  gi, gf, gg, go);
  gemm4_region(dhb_prev + (size_t)mn * DEC_H  + aoff, 1024,
               w0 + 1536, w1 + 1536, w2 + 1536, w3 + 1536, gi, gf, gg, go);
  int j = j0 + mn, mb = hi ? 8 : 0;
  float bi = bd[j], bff = bd[j + 1024], bg = bd[j + 2048], bo = bd[j + 3072];
  for (int r = 0; r < 8; ++r) {
    int m = mb + r;
    float iv = sigm(gi[r] + bi);
    float fv = sigm(gf[r] + bff);
    float gv = tanhf(gg[r] + bg);
    float ov = sigm(go[r] + bo);
    float c2 = fv * dc[m * DEC_H + j] + iv * gv;
    float h2 = ov * tanhf(c2);
    dc[m * DEC_H + j] = c2;
    dhb_new[m * DEC_H + j] = f2bf(h2);
  }
}

// Projection: (16x1536) @ (1536x80), 5 WMMA column tiles, 2 regions.
__global__ __launch_bounds__(160) void k_proj(
    const unsigned short* Wp, const float* pjb,
    const unsigned short* dhb, const unsigned short* ctxb, float* mel, int t) {
  int  n0   = (threadIdx.x >> 5) * 16;     // 0..64
  int  lane = threadIdx.x & 31;
  bool hi   = lane >= 16;
  int  mn   = lane & 15;
  const unsigned short* wr = Wp + (size_t)(n0 + mn) * KP + (hi ? 16 : 0);
  v8f acc = {};
  gemm1_region(dhb  + (size_t)mn * DEC_H + (hi ? 16 : 0), 1024, wr,        acc);
  gemm1_region(ctxb + (size_t)mn * ENC_H + (hi ? 16 : 0), 512,  wr + 1024, acc);
  int   n    = n0 + mn;
  float bias = pjb[n];
  for (int r = 0; r < 8; ++r) {
    int m = (hi ? 8 : 0) + r;
    mel[((size_t)m * T_DEC + t) * N_MELS + n] = acc[r] + bias;
  }
}

// ---------------------------------------------------------------------------

extern "C" void kernel_launch(void* const* d_in, const int* in_sizes, int n_in,
                              void* d_out, int out_size, void* d_ws, size_t ws_size,
                              hipStream_t stream) {
  (void)in_sizes; (void)n_in; (void)out_size; (void)ws_size;
  const float* enc   = (const float*)d_in[0];
  const float* tgt   = (const float*)d_in[1];
  const float* pw    = (const float*)d_in[2];
  const float* pb    = (const float*)d_in[3];
  const float* Mw    = (const float*)d_in[4];
  const float* Qw    = (const float*)d_in[5];
  const float* Ww    = (const float*)d_in[6];
  const float* Lw    = (const float*)d_in[7];
  const float* convw = (const float*)d_in[8];
  const float* awih  = (const float*)d_in[9];
  const float* awhh  = (const float*)d_in[10];
  const float* abih  = (const float*)d_in[11];
  const float* abhh  = (const float*)d_in[12];
  const float* dwih  = (const float*)d_in[13];
  const float* dwhh  = (const float*)d_in[14];
  const float* dbih  = (const float*)d_in[15];
  const float* dbhh  = (const float*)d_in[16];
  const float* pjw   = (const float*)d_in[17];
  const float* pjb   = (const float*)d_in[18];

  char* ws = (char*)d_ws;
  size_t off = 0;
  auto carve = [&](size_t bytes) -> void* {
    off = (off + 255) & ~(size_t)255;
    void* p = ws + off;
    off += bytes;
    return p;
  };
  unsigned short* Wa  = (unsigned short*)carve((size_t)GA * KA * 2);
  unsigned short* Wd  = (unsigned short*)carve((size_t)GA * KD * 2);
  unsigned short* Wp  = (unsigned short*)carve((size_t)N_MELS * KP * 2);
  unsigned short* Qwb = (unsigned short*)carve((size_t)ATTN_D * ATTN_H * 2);
  float* ba  = (float*)carve((size_t)GA * 4);
  float* bd  = (float*)carve((size_t)GA * 4);
  float* pme = (float*)carve((size_t)B_ * T_ENC * ATTN_D * 4);
  unsigned short* pre = (unsigned short*)carve((size_t)T_DEC * B_ * PRENET_ * 2);
  // recurrent state (zeroed each call)
  float* ac  = (float*)carve((size_t)B_ * ATTN_H * 4);
  float* dc  = (float*)carve((size_t)B_ * DEC_H * 4);
  float* aw  = (float*)carve((size_t)B_ * T_ENC * 4);
  float* aws = (float*)carve((size_t)B_ * T_ENC * 4);
  float* qv  = (float*)carve((size_t)B_ * ATTN_D * 4);
  float* ev  = (float*)carve((size_t)B_ * T_ENC * 4);
  unsigned short* ctxb   = (unsigned short*)carve((size_t)B_ * ENC_H * 2);
  unsigned short* ahb[2] = { (unsigned short*)carve((size_t)B_ * ATTN_H * 2),
                             (unsigned short*)carve((size_t)B_ * ATTN_H * 2) };
  unsigned short* dhb[2] = { (unsigned short*)carve((size_t)B_ * DEC_H * 2),
                             (unsigned short*)carve((size_t)B_ * DEC_H * 2) };
  size_t state_bytes = (size_t)((ws + off) - (char*)ac);

  k_prep_weights<<<2048, 256, 0, stream>>>(awih, awhh, abih, abhh,
                                           dwih, dwhh, dbih, dbhh, pjw, Qw,
                                           Wa, Wd, Wp, Qwb, ba, bd);
  k_prenet<<<(T_DEC * B_ * PRENET_ + 255) / 256, 256, 0, stream>>>(tgt, pw, pb, pre);
  k_pme<<<(B_ * T_ENC * ATTN_D + 255) / 256, 256, 0, stream>>>(enc, Mw, pme);
  (void)hipMemsetAsync(ac, 0, state_bytes, stream);

  float* mel_out   = (float*)d_out;                                  // (16,250,80)
  float* align_out = mel_out + (size_t)B_ * T_DEC * N_MELS;          // (16,250,200)

  for (int t = 0; t < T_DEC; ++t) {
    const unsigned short* pre_t = pre + (size_t)t * B_ * PRENET_;
    unsigned short* ahp = ahb[t & 1];
    unsigned short* ahn = ahb[(t + 1) & 1];
    unsigned short* dhp = dhb[t & 1];
    unsigned short* dhn = dhb[(t + 1) & 1];
    k_attn_lstm<<<16, 128, 0, stream>>>(Wa, ba, pre_t, ctxb, ahp, ahn, ac);
    k_q<<<1, 256, 0, stream>>>(Qwb, ahn, qv);
    k_energy<<<B_ * T_ENC, 128, 0, stream>>>(qv, pme, aw, aws, Lw, Ww, convw, ev);
    k_softmax_ctx<<<B_, 256, 0, stream>>>(ev, enc, aw, aws, ctxb, align_out, t);
    k_dec_lstm<<<16, 128, 0, stream>>>(Wd, bd, ctxb, ahn, dhp, dhn, dc);
    k_proj<<<1, 160, 0, stream>>>(Wp, pjb, dhn, ctxb, mel_out, t);
  }
}
